// SpectralConv_39642548142249
// MI455X (gfx1250) — matrix operand order repaired and checked by reading, not compiled, not run on hardware
//
#include <hip/hip_runtime.h>

// SpectralConv (FNO layer) for gfx1250: all spectral transforms expressed as
// partial DFT GEMMs on v_wmma_f32_16x16x32_f16 (f16 in, f32 acc).
// B=16, H=W=256, CIN=COUT=32, modes: kh in {0..15, 240..255}, kw in 0..16.
// All LDS B operands are stored pre-swizzled in the WMMA B-fragment layout so
// every fragment load is one contiguous 32-byte read (2x ds_load_b128).

#define PI_F 3.14159265358979323846f

typedef __attribute__((ext_vector_type(16))) _Float16 v16h;
typedef __attribute__((ext_vector_type(8)))  _Float16 v8h;
typedef __attribute__((ext_vector_type(8)))  float    v8f;
typedef __attribute__((ext_vector_type(8)))  unsigned int v8u;

static __device__ __forceinline__ v8f wmma16(v16h a, v16h b, v8f c) {
  // (neg_a, A, neg_b, B, c_mod, C, reuse_a, reuse_b)
  return __builtin_amdgcn_wmma_f32_16x16x32_f16(false, a, false, b, (short)0, c,
                                                false, false);
}

// Negate all 16 packed halves via sign-bit XOR (8 VALU ops).
static __device__ __forceinline__ v16h hneg(v16h x) {
  v8u u = __builtin_bit_cast(v8u, x);
  u ^= 0x80008000u;
  return __builtin_bit_cast(v16h, u);
}

// A fragment (16x32 MxK f16), row-major source, unit K-stride.
// Lane L: m=L&15, hi=L>>4; halves 0..15 = K hi*16..hi*16+15.
static __device__ __forceinline__ v16h frag_a(const _Float16* p, int ms) {
  const int lane = threadIdx.x & 31;
  const _Float16* q = p + (lane & 15) * ms + ((lane >> 4) << 4);
  v8h lo = *(const v8h*)(q);
  v8h hi = *(const v8h*)(q + 8);
  return __builtin_shufflevector(lo, hi, 0, 1, 2, 3, 4, 5, 6, 7, 8, 9, 10, 11,
                                 12, 13, 14, 15);
}

// Offset (in halfs) of element (k,n) inside a swizzled B buffer tiled as
// (K/32) x ntiles tiles of 512 halfs. Within a 32x16 tile:
// lane = (n&15) + 16*((k>>4)&1), slot = k&15.
static __device__ __forceinline__ int bidx(int k, int n, int ntiles) {
  int kt = k >> 5;
  int lane = (n & 15) + (((k >> 4) & 1) << 4);
  return ((kt * ntiles + (n >> 4)) * 32 + lane) * 16 + (k & 15);
}

// B fragment from swizzled storage: one contiguous 32B read per lane.
static __device__ __forceinline__ v16h frag_b_swz(const _Float16* tile) {
  return *(const v16h*)(tile + ((threadIdx.x & 31) << 4));
}

// ---------------- Kernel 1: forward rfft along W (partial, 17 bins) --------
// out[cin][kw] = sum_w xT[cin][w] * e^{-2pi i kw w/256} / 256
// Xw layout: [(b*256+h)*17 + kw]*64 + cin*2 + {re,im}   (f16)
__global__ __launch_bounds__(256) void k1_fwd_w(const float* __restrict__ x,
                                                _Float16* __restrict__ Xw) {
  __shared__ __align__(32) _Float16 sXT[32 * 256];  // [cin][w]
  __shared__ __align__(32) _Float16 sTc[8 * 2 * 512];  // swz [w=K][kw=N] cos/256
  __shared__ __align__(32) _Float16 sTs[8 * 2 * 512];  // swz -sin/256
  const int bh = blockIdx.x;  // b*256 + h
  const int t = threadIdx.x;
  const float* xp = x + (size_t)bh * (256 * 32);
  for (int i = t; i < 256 * 32; i += 256) {
    int w = i >> 5, c = i & 31;
    sXT[c * 256 + w] = (_Float16)xp[i];
  }
  for (int i = t; i < 256 * 32; i += 256) {
    int w = i >> 5, kw = i & 31;
    int ph = (w * kw) & 255;
    float sv, cv;
    __sincosf((float)ph * (2.0f * PI_F / 256.0f), &sv, &cv);
    int d = bidx(w, kw, 2);
    sTc[d] = (_Float16)(cv * (1.0f / 256.0f));
    sTs[d] = (_Float16)(-sv * (1.0f / 256.0f));
  }
  __syncthreads();
  const int wave = t >> 5;
  const int mt = (wave >> 2) & 1;  // cin tile
  const int nt = (wave >> 1) & 1;  // kw tile
  const int part = wave & 1;       // 0=re, 1=im
  const _Float16* T = part ? sTs : sTc;
  v8f acc = {};
#pragma unroll
  for (int ks = 0; ks < 8; ++ks) {
    v16h a = frag_a(sXT + mt * 16 * 256 + ks * 32, 256);
    v16h b = frag_b_swz(T + (ks * 2 + nt) * 512);
    acc = wmma16(a, b, acc);
  }
  const int lane = t & 31;
  const int n = nt * 16 + (lane & 15);  // kw
  if (n < 17) {
    const int mb = mt * 16 + ((lane >> 4) * 8);
#pragma unroll
    for (int r = 0; r < 8; ++r) {
      int cin = mb + r;
      Xw[((size_t)bh * 17 + n) * 64 + cin * 2 + part] = (_Float16)acc[r];
    }
  }
}

// ---------------- Kernel 2: forward FFT along H (32 kept rows) -------------
// Xh[kh][kw,cin] = sum_h (cos - i sin)/256 * Xw[h][kw,cin]
// Xh layout (mode-major): [(kh*17+kw)*16 + b]*64 + cin*2 + {re,im}
__global__ __launch_bounds__(256) void k2_fwd_h(const _Float16* __restrict__ Xw,
                                                _Float16* __restrict__ Xh) {
  __shared__ __align__(32) _Float16 sAc[32 * 256];     // [kh][h]  cos/256
  __shared__ __align__(32) _Float16 sAs[32 * 256];     // [kh][h] -sin/256
  __shared__ __align__(32) _Float16 sBr[8 * 2 * 512];  // swz [h=K][nc=N] re
  __shared__ __align__(32) _Float16 sBi[8 * 2 * 512];  // swz im
  const int b = blockIdx.x / 17;
  const int np = blockIdx.x % 17;  // panel of 32 n-columns
  const int t = threadIdx.x;
  for (int i = t; i < 32 * 256; i += 256) {
    int kh = i >> 8, h = i & 255;
    int khp = kh < 16 ? kh : kh + 224;  // rows 0..15 and 240..255
    int ph = (khp * h) & 255;
    float sv, cv;
    __sincosf((float)ph * (2.0f * PI_F / 256.0f), &sv, &cv);
    sAc[i] = (_Float16)(cv * (1.0f / 256.0f));
    sAs[i] = (_Float16)(-sv * (1.0f / 256.0f));
  }
  {
    const unsigned int* src = (const unsigned int*)Xw;
    for (int i = t; i < 256 * 32; i += 256) {
      int h = i >> 5, nc = i & 31;
      int ncol = np * 32 + nc;
      int kw = ncol >> 5, cin = ncol & 31;
      unsigned int v = src[((size_t)(b * 256 + h) * 17 + kw) * 32 + cin];
      int d = bidx(h, nc, 2);
      sBr[d] = __builtin_bit_cast(_Float16, (unsigned short)(v & 0xffffu));
      sBi[d] = __builtin_bit_cast(_Float16, (unsigned short)(v >> 16));
    }
  }
  __syncthreads();
  const int wave = t >> 5;
  const int nt = wave & 1, part = (wave >> 1) & 1, mt = (wave >> 2) & 1;
  v8f acc = {};
#pragma unroll
  for (int ks = 0; ks < 8; ++ks) {
    v16h ac = frag_a(sAc + mt * 16 * 256 + ks * 32, 256);
    v16h as = frag_a(sAs + mt * 16 * 256 + ks * 32, 256);
    v16h br = frag_b_swz(sBr + (ks * 2 + nt) * 512);
    v16h bi = frag_b_swz(sBi + (ks * 2 + nt) * 512);
    if (part == 0) {  // Re = Ac@Br + As@(-Bi)
      acc = wmma16(ac, br, acc);
      acc = wmma16(as, hneg(bi), acc);
    } else {  // Im = Ac@Bi + As@Br
      acc = wmma16(ac, bi, acc);
      acc = wmma16(as, br, acc);
    }
  }
  const int lane = t & 31;
  const int nc = nt * 16 + (lane & 15);
  const int ncol = np * 32 + nc;
  const int kw = ncol >> 5, cin = ncol & 31;
  const int mb = mt * 16 + ((lane >> 4) * 8);
#pragma unroll
  for (int r = 0; r < 8; ++r) {
    int kh = mb + r;
    Xh[(((size_t)kh * 17 + kw) * 16 + b) * 64 + cin * 2 + part] =
        (_Float16)acc[r];
  }
}

// ---------------- Kernel 3: per-mode complex channel mixing ----------------
// Y[b][o] = sum_i X[b][i] * W[i][o]  (complex), one mode per block.
// Ym layout: [((b*32+kh)*17+kw)]*64 + o*2 + {re,im}
__global__ __launch_bounds__(128) void k3_mix(
    const _Float16* __restrict__ Xh, const float* __restrict__ w1r,
    const float* __restrict__ w1i, const float* __restrict__ w2r,
    const float* __restrict__ w2i, _Float16* __restrict__ Ym) {
  __shared__ __align__(32) _Float16 sXr[16 * 32];  // [b][i]
  __shared__ __align__(32) _Float16 sXi[16 * 32];
  __shared__ __align__(32) _Float16 sWr[2 * 512];  // swz [i=K][o=N]
  __shared__ __align__(32) _Float16 sWi[2 * 512];
  const int mode = blockIdx.x;  // kh*17 + kw
  const int kh = mode / 17, kw = mode % 17;
  const float* wr = kh < 16 ? w1r : w2r;
  const float* wi = kh < 16 ? w1i : w2i;
  const int khl = kh & 15;
  const int t = threadIdx.x;
  for (int i = t; i < 1024; i += 128) {
    int ci = i >> 5, o = i & 31;  // weights: (CIN,COUT,EM0,EM1)
    size_t gi = ((size_t)(ci * 32 + o) * 16 + khl) * 17 + kw;
    int d = bidx(ci, o, 2);
    sWr[d] = (_Float16)wr[gi];
    sWi[d] = (_Float16)wi[gi];
  }
  {
    const unsigned int* src = (const unsigned int*)Xh;
    for (int i = t; i < 512; i += 128) {  // 16 b x 32 i complex pairs
      unsigned int v = src[(size_t)mode * 512 + i];
      sXr[i] = __builtin_bit_cast(_Float16, (unsigned short)(v & 0xffffu));
      sXi[i] = __builtin_bit_cast(_Float16, (unsigned short)(v >> 16));
    }
  }
  __syncthreads();
  const int wave = t >> 5;
  const int nt = wave & 1, part = (wave >> 1) & 1;
  v16h ar = frag_a(sXr, 32);
  v16h ai = frag_a(sXi, 32);
  v16h br = frag_b_swz(sWr + nt * 512);
  v16h bi = frag_b_swz(sWi + nt * 512);
  v8f acc = {};
  if (part == 0) {  // Re = Xr@Wr + Xi@(-Wi)
    acc = wmma16(ar, br, acc);
    acc = wmma16(ai, hneg(bi), acc);
  } else {  // Im = Xr@Wi + Xi@Wr
    acc = wmma16(ar, bi, acc);
    acc = wmma16(ai, br, acc);
  }
  const int lane = t & 31;
  const int o = nt * 16 + (lane & 15);
  const int mb = (lane >> 4) * 8;
#pragma unroll
  for (int r = 0; r < 8; ++r) {
    int bb = mb + r;  // batch (M=16)
    Ym[(((size_t)bb * 32 + kh) * 17 + kw) * 64 + o * 2 + part] =
        (_Float16)acc[r];
  }
}

// ---------------- Kernel 4: inverse FFT along H ----------------------------
// Z[h][kw,o] = sum_kh (cos + i sin) * Ym[kh][kw,o]   (unnormalized inverse)
// Z layout: [((b*256+h)*17+kw)*2 + ri]*32 + o
__global__ __launch_bounds__(256) void k4_inv_h(const _Float16* __restrict__ Ym,
                                                _Float16* __restrict__ Z) {
  __shared__ __align__(32) _Float16 sAc[256 * 32];  // [h][kh] cos
  __shared__ __align__(32) _Float16 sAs[256 * 32];  // [h][kh] sin
  __shared__ __align__(32) _Float16 sBr[2 * 512];   // swz [kh=K][nc=N]
  __shared__ __align__(32) _Float16 sBi[2 * 512];
  const int b = blockIdx.x / 17;
  const int np = blockIdx.x % 17;
  const int t = threadIdx.x;
  for (int i = t; i < 256 * 32; i += 256) {
    int h = i >> 5, kh = i & 31;
    int khp = kh < 16 ? kh : kh + 224;
    int ph = (khp * h) & 255;
    float sv, cv;
    __sincosf((float)ph * (2.0f * PI_F / 256.0f), &sv, &cv);
    sAc[i] = (_Float16)cv;
    sAs[i] = (_Float16)sv;
  }
  {
    const unsigned int* src = (const unsigned int*)Ym;
    for (int i = t; i < 32 * 32; i += 256) {
      int kh = i >> 5, nc = i & 31;
      int ncol = np * 32 + nc;
      int kw = ncol >> 5, o = ncol & 31;
      unsigned int v = src[((size_t)(b * 32 + kh) * 17 + kw) * 32 + o];
      int d = bidx(kh, nc, 2);
      sBr[d] = __builtin_bit_cast(_Float16, (unsigned short)(v & 0xffffu));
      sBi[d] = __builtin_bit_cast(_Float16, (unsigned short)(v >> 16));
    }
  }
  __syncthreads();
  const int wave = t >> 5;
  const int nt = wave & 1, part = (wave >> 1) & 1, mg = (wave >> 2) & 1;
  v16h br = frag_b_swz(sBr + nt * 512);
  v16h bi = frag_b_swz(sBi + nt * 512);
  v16h bin = hneg(bi);
  for (int mi = 0; mi < 8; ++mi) {
    int mt = mg * 8 + mi;  // h tile (16 total)
    v16h ac = frag_a(sAc + mt * 16 * 32, 32);
    v16h as = frag_a(sAs + mt * 16 * 32, 32);
    v8f acc = {};
    if (part == 0) {  // Re = cos@Or + sin@(-Oi)
      acc = wmma16(ac, br, acc);
      acc = wmma16(as, bin, acc);
    } else {  // Im = cos@Oi + sin@Or
      acc = wmma16(ac, bi, acc);
      acc = wmma16(as, br, acc);
    }
    const int lane = t & 31;
    const int nc = nt * 16 + (lane & 15);
    const int ncol = np * 32 + nc;
    const int kw = ncol >> 5, o = ncol & 31;
    const int mb = mt * 16 + ((lane >> 4) * 8);
#pragma unroll
    for (int r = 0; r < 8; ++r) {
      int h = mb + r;
      Z[(((size_t)(b * 256 + h) * 17 + kw) * 2 + part) * 32 + o] =
          (_Float16)acc[r];
    }
  }
}

// ---------------- Kernel 5: inverse rfft along W + bias --------------------
// y[w][o] = sum_j Twi[w][j] * Zp[j][o],  j packs (kw, re/im), K padded 34->64.
// Twi[w][(kw,re)] = c_kw cos(2pi kw w/256); [(kw,im)] = -c_kw sin(...);
// c_0=1 (imag of bin0 auto-dropped since sin(0)=0), c_kw=2 for kw>=1.
__global__ __launch_bounds__(256) void k5_inv_w(const _Float16* __restrict__ Z,
                                                const float* __restrict__ bias,
                                                float* __restrict__ y) {
  __shared__ __align__(32) _Float16 sT[256 * 64];     // [w][j]
  __shared__ __align__(32) _Float16 sB[2 * 2 * 512];  // swz [j=K][o=N]
  const int bh = blockIdx.x;  // b*256 + h
  const int t = threadIdx.x;
  for (int i = t; i < 256 * 64; i += 256) {
    int w = i >> 6, j = i & 63;
    int kw = j >> 1, ri = j & 1;
    _Float16 v = (_Float16)0.0f;
    if (kw < 17) {
      int ph = (kw * w) & 255;
      float sv, cv;
      __sincosf((float)ph * (2.0f * PI_F / 256.0f), &sv, &cv);
      float c = (kw == 0) ? 1.0f : 2.0f;
      v = (_Float16)(ri == 0 ? c * cv : -c * sv);
    }
    sT[i] = v;
  }
  for (int i = t; i < 64 * 32; i += 256) {
    int j = i >> 5, o = i & 31;
    _Float16 v = (_Float16)0.0f;
    if (j < 34) v = Z[(size_t)bh * 1088 + j * 32 + o];
    sB[bidx(j, o, 2)] = v;
  }
  __syncthreads();
  const int wave = t >> 5;
  const int nt = wave & 1, mg = wave >> 1;  // mg 0..3
  v16h b0 = frag_b_swz(sB + (0 * 2 + nt) * 512);
  v16h b1 = frag_b_swz(sB + (1 * 2 + nt) * 512);
  for (int mi = 0; mi < 4; ++mi) {
    int mt = mg * 4 + mi;  // w tile (16 total)
    v8f acc = {};
    acc = wmma16(frag_a(sT + mt * 16 * 64 + 0 * 32, 64), b0, acc);
    acc = wmma16(frag_a(sT + mt * 16 * 64 + 1 * 32, 64), b1, acc);
    const int lane = t & 31;
    const int o = nt * 16 + (lane & 15);
    const float bv = bias[o];
    const int mb = mt * 16 + ((lane >> 4) * 8);
#pragma unroll
    for (int r = 0; r < 8; ++r) {
      int w = mb + r;
      y[((size_t)bh * 256 + w) * 32 + o] = acc[r] + bv;
    }
  }
}

extern "C" void kernel_launch(void* const* d_in, const int* in_sizes, int n_in,
                              void* d_out, int out_size, void* d_ws,
                              size_t ws_size, hipStream_t stream) {
  (void)in_sizes; (void)n_in; (void)out_size;
  const float* x = (const float*)d_in[0];
  const float* w1r = (const float*)d_in[1];
  const float* w1i = (const float*)d_in[2];
  const float* w2r = (const float*)d_in[3];
  const float* w2i = (const float*)d_in[4];
  const float* bias = (const float*)d_in[5];
  float* y = (float*)d_out;

  // Workspace (f16 halves): ~20.1 MB total, L2-resident.
  const size_t XW = (size_t)16 * 256 * 17 * 32 * 2;  // 4456448
  const size_t XH = (size_t)32 * 17 * 16 * 32 * 2;   // 557056
  const size_t YM = XH;
  (void)ws_size;  // requires ~20.1 MB
  _Float16* Xw = (_Float16*)d_ws;
  _Float16* Xh = Xw + XW;
  _Float16* Ym = Xh + XH;
  _Float16* Z = Ym + YM;

  k1_fwd_w<<<16 * 256, 256, 0, stream>>>(x, Xw);
  k2_fwd_h<<<16 * 17, 256, 0, stream>>>(Xw, Xh);
  k3_mix<<<32 * 17, 128, 0, stream>>>(Xh, w1r, w1i, w2r, w2i, Ym);
  k4_inv_h<<<16 * 17, 256, 0, stream>>>(Ym, Z);
  k5_inv_w<<<16 * 256, 256, 0, stream>>>(Z, bias, y);
}
